// SO2EquivariantGraphAttention_83485574299846
// MI455X (gfx1250) — matrix operand order repaired and checked by reading, not compiled, not run on hardware
//
#include <hip/hip_runtime.h>

typedef _Float16 f16;
typedef __attribute__((ext_vector_type(16))) _Float16 v16h;
typedef __attribute__((ext_vector_type(4)))  _Float16 v4h;
typedef __attribute__((ext_vector_type(8)))  float    v8f;
typedef __attribute__((ext_vector_type(4)))  unsigned  v4u;

union V16 { v4u u[2]; v16h h; };

__device__ __forceinline__ float sigm(float x) { return 1.f / (1.f + __expf(-x)); }
__device__ __forceinline__ unsigned encF(float f) {
  unsigned u = __float_as_uint(f);
  return (u & 0x80000000u) ? ~u : (u | 0x80000000u);
}
__device__ __forceinline__ float decF(unsigned e) {
  unsigned u = (e & 0x80000000u) ? (e ^ 0x80000000u) : ~e;
  return __uint_as_float(u);
}

// ---------------------------------------------------------------- utilities
__global__ void zero32(unsigned* p, long n) {
  long i = (long)blockIdx.x * 256 + threadIdx.x;
  if (i < n) p[i] = 0u;
}

__global__ void cvt_f32_f16(const float* __restrict__ in, f16* __restrict__ out, long n) {
  long i = (long)blockIdx.x * 256 + threadIdx.x;
  if (i < n) out[i] = (f16)in[i];
}

// transpose+convert: in (K x Nin) f32 row-major -> out (Nout x K) f16, cols n0..n0+Nout-1
__global__ void tcvt(const float* __restrict__ in, f16* __restrict__ out,
                     int K, int Nin, int n0, int Nout) {
  long i = (long)blockIdx.x * 256 + threadIdx.x;
  if (i >= (long)K * Nout) return;
  int n = (int)(i / K), k = (int)(i % K);
  out[(long)n * K + k] = (f16)in[(long)k * Nin + n0 + n];
}

// B' = [[Wr, Wi], [-Wi, Wr]] for conv2 m=1: w1 (128x512) -> out transposed (512 x 256)
__global__ void build_c2b1(const float* __restrict__ w1, f16* __restrict__ out) {
  long i = (long)blockIdx.x * 256 + threadIdx.x;
  if (i >= 512L * 256) return;
  int n = (int)(i / 256), k = (int)(i % 256);
  float v;
  if (k < 128)      v = w1[(long)k * 512 + n];
  else if (n < 256) v = -w1[(long)(k - 128) * 512 + (n + 256)];
  else              v = w1[(long)(k - 128) * 512 + (n - 256)];
  out[i] = (f16)v;
}

// conv2 m=2: w2 (64x256) -> out transposed (256 x 128)
__global__ void build_c2b2(const float* __restrict__ w2, f16* __restrict__ out) {
  long i = (long)blockIdx.x * 256 + threadIdx.x;
  if (i >= 256L * 128) return;
  int n = (int)(i / 128), k = (int)(i % 128);
  float v;
  if (k < 64)       v = w2[(long)k * 256 + n];
  else if (n < 128) v = -w2[(long)(k - 64) * 256 + (n + 128)];
  else              v = w2[(long)(k - 64) * 256 + (n - 128)];
  out[i] = (f16)v;
}

// x_edge (E,192) f16 = [edge_distance | src_emb[an[src]] | tgt_emb[an[dst]]]
__global__ void build_xedge(const float* __restrict__ ed, const float* __restrict__ semb,
                            const float* __restrict__ temb, const int* __restrict__ an,
                            const int* __restrict__ srcI, const int* __restrict__ dstI,
                            f16* __restrict__ xe, long total) {
  long i = (long)blockIdx.x * 256 + threadIdx.x;
  if (i >= total) return;
  long e = i / 192; int c = (int)(i % 192);
  float v;
  if (c < 64)       v = ed[e * 64 + c];
  else if (c < 128) v = semb[(long)an[srcI[e]] * 64 + (c - 64)];
  else              v = temb[(long)an[dstI[e]] * 64 + (c - 128)];
  xe[i] = (f16)v;
}

// ---------------------------------------------------------------- WMMA GEMM
// C(MxN) = A(MxK) * B(KxN) + bias ; A f16 row-major (stride lda),
// B pre-transposed f16: B[n*K + k].
// Wave tile: 16 rows x 64 cols (4 accumulator chains, 1 A fetch per K-step
// amortized over 4 WMMAs). Block = 8 waves on 8 distinct m-tiles -> 128x64.
// mode 0: f32 out[row*ldd+col]; mode 1: f16 out[row*ldd+col];
// mode 2: f16 vmsg[row*1152 + frow*128 + (col&127)], frow = nibble of rowmap.
__global__ __launch_bounds__(256) void gemm_wmma(
    const f16* __restrict__ A, int lda, const f16* __restrict__ B,
    const float* __restrict__ bias, void* __restrict__ out,
    int M, int K, int mode, int ldd, int rowmap) {
  const int lane = threadIdx.x & 31;
  const int wave = threadIdx.x >> 5;
  const int row0 = (blockIdx.x * 8 + wave) * 16;
  const int col0 = blockIdx.y * 64;
  const int l15 = lane & 15;
  const int hf  = lane >> 4;

  int arow = row0 + l15; if (arow > M - 1) arow = M - 1;
  const f16* Ap  = A + (long)arow * lda + 8 * hf;
  const f16* Bp0 = B + (long)(col0 + l15) * K + 8 * hf;
  const long bstep = (long)16 * K;

  v8f acc0 = {0.f, 0.f, 0.f, 0.f, 0.f, 0.f, 0.f, 0.f};
  v8f acc1 = acc0, acc2 = acc0, acc3 = acc0;
  for (int k = 0; k < K; k += 32) {
    V16 a, b0, b1, b2, b3;
    a.u[0]  = *(const v4u*)(Ap + k);        // A lanes: K = k + e + 8*hf (e<8)
    a.u[1]  = *(const v4u*)(Ap + k + 16);   //          K = k + 16 + (e-8) + 8*hf
    b0.u[0] = *(const v4u*)(Bp0 + k);       // B lanes: same K order as A, N = lane&15
    b0.u[1] = *(const v4u*)(Bp0 + k + 16);
    b1.u[0] = *(const v4u*)(Bp0 + bstep + k);
    b1.u[1] = *(const v4u*)(Bp0 + bstep + k + 16);
    b2.u[0] = *(const v4u*)(Bp0 + 2 * bstep + k);
    b2.u[1] = *(const v4u*)(Bp0 + 2 * bstep + k + 16);
    b3.u[0] = *(const v4u*)(Bp0 + 3 * bstep + k);
    b3.u[1] = *(const v4u*)(Bp0 + 3 * bstep + k + 16);
    acc0 = __builtin_amdgcn_wmma_f32_16x16x32_f16(false, a.h, false, b0.h, (short)0, acc0, false, false);
    acc1 = __builtin_amdgcn_wmma_f32_16x16x32_f16(false, a.h, false, b1.h, (short)0, acc1, false, false);
    acc2 = __builtin_amdgcn_wmma_f32_16x16x32_f16(false, a.h, false, b2.h, (short)0, acc2, false, false);
    acc3 = __builtin_amdgcn_wmma_f32_16x16x32_f16(false, a.h, false, b3.h, (short)0, acc3, false, false);
  }
#pragma unroll
  for (int t = 0; t < 4; ++t) {
    v8f acc = (t == 0) ? acc0 : (t == 1) ? acc1 : (t == 2) ? acc2 : acc3;
    int col = col0 + l15 + t * 16;
    float bv = bias ? bias[col] : 0.f;
#pragma unroll
    for (int r = 0; r < 8; ++r) {
      int row = row0 + 8 * hf + r;
      if (row >= M) continue;
      float v = acc[r] + bv;
      if (mode == 0) {
        ((float*)out)[(long)row * ldd + col] = v;
      } else if (mode == 1) {
        ((f16*)out)[(long)row * ldd + col] = (f16)v;
      } else {
        int frow = (rowmap >> ((col >> 7) * 4)) & 15;
        ((f16*)out)[(long)row * 1152 + frow * 128 + (col & 127)] = (f16)v;
      }
    }
  }
}

// ------------------------------------------------ LayerNorm(64) + SiLU -> f16
__global__ __launch_bounds__(256) void ln_silu(const float* __restrict__ in,
                                               const float* __restrict__ g,
                                               const float* __restrict__ b,
                                               f16* __restrict__ out, int M) {
  int lane = threadIdx.x & 31;
  long row = (long)blockIdx.x * 8 + (threadIdx.x >> 5);
  if (row >= M) return;
  const float* p = in + row * 64;
  float x0 = p[lane], x1 = p[lane + 32];
  float s = x0 + x1, q = x0 * x0 + x1 * x1;
#pragma unroll
  for (int m = 16; m >= 1; m >>= 1) { s += __shfl_xor(s, m, 32); q += __shfl_xor(q, m, 32); }
  float mu = s * (1.f / 64.f);
  float var = q * (1.f / 64.f) - mu * mu;
  float rs = rsqrtf(var + 1e-5f);
  float y0 = (x0 - mu) * rs * g[lane] + b[lane];
  float y1 = (x1 - mu) * rs * g[lane + 32] + b[lane + 32];
  out[row * 64 + lane]      = (f16)(y0 * sigm(y0));
  out[row * 64 + lane + 32] = (f16)(y1 * sigm(y1));
}

// ------------- x0 (E,384) f16: m=0 rows (0,2,6) of Wigner-rotated [xs|xt], * r
__global__ __launch_bounds__(256) void build_x0(const float* __restrict__ x,
                                                const int* __restrict__ srcI,
                                                const int* __restrict__ dstI,
                                                const float* __restrict__ W,
                                                const f16* __restrict__ r,
                                                f16* __restrict__ x0, int E_) {
  int lane = threadIdx.x & 31;
  long e = (long)blockIdx.x * 8 + (threadIdx.x >> 5);
  if (e >= E_) return;
  int s = srcI[e], d = dstI[e];
  const float* We = W + e * 81;
  const f16* re = r + e * 384;
  f16* xo = x0 + e * 384;
#pragma unroll
  for (int t = 0; t < 4; ++t) {
    int ch = lane + 32 * t;                     // channel of concat(xs, xt) (128)
    const float* xp = (ch < 64) ? (x + (long)s * 576 + ch)
                                : (x + (long)d * 576 + (ch - 64));
    float v[9];
#pragma unroll
    for (int j = 0; j < 9; ++j) v[j] = xp[j * 64];
    float rot0 = We[0] * v[0];                                          // (l=0,m=0)
    float rot2 = We[19] * v[1] + We[20] * v[2] + We[21] * v[3];         // row 2, j=1..3
    float rot6 = We[58] * v[4] + We[59] * v[5] + We[60] * v[6]          // row 6, j=4..8
               + We[61] * v[7] + We[62] * v[8];
    xo[ch]       = (f16)(rot0 * (float)re[ch]);
    xo[128 + ch] = (f16)(rot2 * (float)re[128 + ch]);
    xo[256 + ch] = (f16)(rot6 * (float)re[256 + ch]);
  }
}

// ------- gating -> msg2 in PERM order (E,9,64) f16 ; alpha (E,8) + atomicMax amax
__global__ __launch_bounds__(256) void gate_alpha(
    const float* __restrict__ x0e, const float* __restrict__ x,
    const int* __restrict__ dstI, const float* __restrict__ rl,
    const float* __restrict__ alng, const float* __restrict__ alnb,
    const float* __restrict__ adot, f16* __restrict__ msg2,
    float* __restrict__ alpha, unsigned* __restrict__ amaxI, int E_) {
  int lane = threadIdx.x & 31;
  long e = (long)blockIdx.x * 8 + (threadIdx.x >> 5);
  if (e >= E_) return;
  const float* xe = x0e + e * 576;
  int d = dstI[e];
  const float* xt = x + (long)d * 576;
  f16* mo = msg2 + e * 576;
  const int INVP[9] = {0, 5, 1, 3, 8, 6, 2, 4, 7};  // emb row i -> perm position
#pragma unroll
  for (int t = 0; t < 2; ++t) {
    int c = lane + 32 * t;
    float os = xe[256 + c];
    mo[INVP[0] * 64 + c] = (f16)(os * sigm(os));                 // silu(o_s)
    float gd1 = sigm(xe[320 + c]), gt1 = sigm(xe[448 + c]);      // l=1 gates
    float gd2 = sigm(xe[384 + c]), gt2 = sigm(xe[512 + c]);      // l=2 gates
#pragma unroll
    for (int j = 1; j <= 3; ++j)
      mo[INVP[j] * 64 + c] = (f16)(gd1 * rl[e * 8 + (j - 1)] + gt1 * xt[j * 64 + c]);
#pragma unroll
    for (int j = 4; j <= 8; ++j)
      mo[INVP[j] * 64 + c] = (f16)(gd2 * rl[e * 8 + (j - 1)] + gt2 * xt[j * 64 + c]);
  }
  // alpha: LN(32) per head + gated act + dot(alpha_dot)
#pragma unroll
  for (int h = 0; h < 8; ++h) {
    float v = xe[h * 32 + lane];
    float s = v, q = v * v;
#pragma unroll
    for (int m = 16; m >= 1; m >>= 1) { s += __shfl_xor(s, m, 32); q += __shfl_xor(q, m, 32); }
    float mu = s * (1.f / 32.f);
    float var = q * (1.f / 32.f) - mu * mu;
    float a = (v - mu) * rsqrtf(var + 1e-5f) * alng[lane] + alnb[lane];
    float act = 0.6f * a + 0.4f * a * (2.f * sigm(a) - 1.f);
    float p = act * adot[h * 32 + lane];
#pragma unroll
    for (int m = 16; m >= 1; m >>= 1) p += __shfl_xor(p, m, 32);
    if (lane == 0) {
      alpha[e * 8 + h] = p;
      atomicMax(&amaxI[(long)d * 8 + h], encF(p));
    }
  }
}

__global__ void soft_exp(float* __restrict__ alpha, const int* __restrict__ dstI,
                         const unsigned* __restrict__ amaxI, float* __restrict__ den, long n) {
  long i = (long)blockIdx.x * 256 + threadIdx.x;
  if (i >= n) return;
  long e = i >> 3; int h = (int)(i & 7);
  long d = dstI[e];
  float ex = __expf(alpha[i] - decF(amaxI[d * 8 + h]));
  alpha[i] = ex;
  atomicAdd(&den[d * 8 + h], ex);
}

// ------- attn: scale by softmax, rotate by W^T (block-diag), scatter-add to node
__global__ __launch_bounds__(256) void attn_scatter(
    const f16* __restrict__ vmsg, const float* __restrict__ alpha,
    const float* __restrict__ den, const float* __restrict__ W,
    const int* __restrict__ dstI, float* __restrict__ node, int E_) {
  int lane = threadIdx.x & 31;
  long e = (long)blockIdx.x * 8 + (threadIdx.x >> 5);
  if (e >= E_) return;
  int d = dstI[e];
  int c = lane * 4;                 // 4 channels/lane; all in one head (h = c>>4)
  int h = lane >> 2;
  float sc = alpha[e * 8 + h] / (den[(long)d * 8 + h] + 1e-16f);
  const f16* vm = vmsg + e * 1152 + c;
  const float* We = W + e * 81;
  float acc[9][4];
#pragma unroll
  for (int i = 0; i < 9; ++i)
#pragma unroll
    for (int k = 0; k < 4; ++k) acc[i][k] = 0.f;
#pragma unroll
  for (int j = 0; j < 9; ++j) {
    v4h av = *(const v4h*)(vm + j * 128);
    float a0 = (float)av[0] * sc, a1 = (float)av[1] * sc;
    float a2 = (float)av[2] * sc, a3 = (float)av[3] * sc;
    int ilo = (j == 0) ? 0 : (j < 4 ? 1 : 4);
    int ihi = (j == 0) ? 0 : (j < 4 ? 3 : 8);
    for (int i = ilo; i <= ihi; ++i) {
      float w = We[j * 9 + i];      // out[i] += W[j,i] * attn[j]
      acc[i][0] += w * a0; acc[i][1] += w * a1;
      acc[i][2] += w * a2; acc[i][3] += w * a3;
    }
  }
  float* nd = node + (long)d * 1152 + c;
#pragma unroll
  for (int i = 0; i < 9; ++i) {
    atomicAdd(nd + i * 128 + 0, acc[i][0]);
    atomicAdd(nd + i * 128 + 1, acc[i][1]);
    atomicAdd(nd + i * 128 + 2, acc[i][2]);
    atomicAdd(nd + i * 128 + 3, acc[i][3]);
  }
}

// ================================================================ host side
extern "C" void kernel_launch(void* const* d_in, const int* in_sizes, int n_in,
                              void* d_out, int out_size, void* d_ws, size_t ws_size,
                              hipStream_t stream) {
  const float* x     = (const float*)d_in[0];
  const int*   an    = (const int*)  d_in[1];
  const float* edist = (const float*)d_in[2];
  const int*   eidx  = (const int*)  d_in[3];
  const float* rl    = (const float*)d_in[4];
  const float* wig   = (const float*)d_in[5];
  const float* semb  = (const float*)d_in[6];
  const float* temb  = (const float*)d_in[7];
  const float* rw1   = (const float*)d_in[8];
  const float* rb1   = (const float*)d_in[9];
  const float* rg1   = (const float*)d_in[10];
  const float* rbt1  = (const float*)d_in[11];
  const float* rw2   = (const float*)d_in[12];
  const float* rb2   = (const float*)d_in[13];
  const float* rg2   = (const float*)d_in[14];
  const float* rbt2  = (const float*)d_in[15];
  const float* rw3   = (const float*)d_in[16];
  const float* rb3   = (const float*)d_in[17];
  const float* c1w0  = (const float*)d_in[18];
  const float* c1b0  = (const float*)d_in[19];
  // d_in[20] (c1_w1), d_in[21] (c1_w2): dead in the reference dataflow
  const float* c2w0  = (const float*)d_in[22];
  const float* c2b0  = (const float*)d_in[23];
  const float* c2w1  = (const float*)d_in[24];
  const float* c2w2  = (const float*)d_in[25];
  const float* alng  = (const float*)d_in[26];
  const float* alnb  = (const float*)d_in[27];
  const float* adot  = (const float*)d_in[28];
  const float* pw    = (const float*)d_in[29];
  const float* pb    = (const float*)d_in[30];

  const long E  = (long)in_sizes[2] / 64;    // edge_distance (E,64)
  const long Nn = (long)in_sizes[0] / 576;   // x (N,9,64)
  const int* srcI = eidx;
  const int* dstI = eidx + E;

  size_t off = 0;
  char* wsb = (char*)d_ws;
  auto carve = [&](size_t bytes) -> char* {
    char* p = wsb + off;
    off += (bytes + 255) & ~(size_t)255;
    return p;
  };
  f16*      xe_x0 = (f16*)  carve((size_t)E * 384 * 2); // x_edge(E,192) then x0(E,384)
  float*    t1    = (float*)carve((size_t)E * 64 * 4);  // also t2
  f16*      t1h   = (f16*)  carve((size_t)E * 64 * 2);  // also t2h
  f16*      rbuf  = (f16*)  carve((size_t)E * 384 * 2); // r ; later nodeh
  float*    x0e   = (float*)carve((size_t)E * 576 * 4); // later vmsg (f16, same bytes)
  char*     mn    =         carve((size_t)E * 576 * 2); // msg2perm f16; later node f32
  float*    alphB = (float*)carve((size_t)E * 8 * 4);
  unsigned* amaxI = (unsigned*)carve((size_t)Nn * 8 * 4);
  float*    denB  = (float*)carve((size_t)Nn * 8 * 4);
  f16* rad1T = (f16*)carve(64  * 192 * 2);
  f16* rad2T = (f16*)carve(64  * 64  * 2);
  f16* rad3T = (f16*)carve(384 * 64  * 2);
  f16* c1w0T = (f16*)carve(576 * 384 * 2);
  f16* c2w0T = (f16*)carve(384 * 192 * 2);
  f16* c2b1T = (f16*)carve(512 * 256 * 2);
  f16* c2b2T = (f16*)carve(256 * 128 * 2);
  f16* projT = (f16*)carve(3 * 64 * 128 * 2);
  (void)ws_size; (void)n_in; (void)out_size;

  f16*   xeb   = xe_x0;
  f16*   x0b   = xe_x0;
  f16*   msg2  = (f16*)mn;
  float* node  = (float*)mn;       // Nn*1152*4 <= E*576*2
  f16*   vmsg  = (f16*)x0e;        // E*1152*2 == E*576*4
  f16*   nodeh = rbuf;             // Nn*1152*2 <= E*384*2
  float* outp  = (float*)d_out;

  dim3 blk(256);
  auto cdiv = [](long a, long b) -> unsigned { return (unsigned)((a + b - 1) / b); };
  auto gemmL = [&](const f16* A, int lda, const f16* B, const float* bias, void* out,
                   long M, int K, int Ncols, int mode, int ldd, int rowmap) {
    dim3 g(cdiv(M, 128), (unsigned)(Ncols / 64), 1);
    gemm_wmma<<<g, blk, 0, stream>>>(A, lda, B, bias, out, (int)M, K, mode, ldd, rowmap);
  };

  // init segment-softmax state
  zero32<<<cdiv(Nn * 8, 256), blk, 0, stream>>>(amaxI, Nn * 8);
  zero32<<<cdiv(Nn * 8, 256), blk, 0, stream>>>((unsigned*)denB, Nn * 8);

  // weight prep (f32 -> f16, N-major for WMMA B operand)
  tcvt<<<cdiv(192L * 64, 256), blk, 0, stream>>>(rw1, rad1T, 192, 64, 0, 64);
  tcvt<<<cdiv(64L * 64, 256), blk, 0, stream>>>(rw2, rad2T, 64, 64, 0, 64);
  tcvt<<<cdiv(64L * 384, 256), blk, 0, stream>>>(rw3, rad3T, 64, 768, 0, 384);
  tcvt<<<cdiv(384L * 576, 256), blk, 0, stream>>>(c1w0, c1w0T, 384, 768, 0, 576);
  tcvt<<<cdiv(192L * 384, 256), blk, 0, stream>>>(c2w0, c2w0T, 192, 384, 0, 384);
  for (int l = 0; l < 3; ++l)
    tcvt<<<cdiv(128L * 64, 256), blk, 0, stream>>>(pw + (long)l * 8192, projT + (long)l * 8192,
                                                   128, 64, 0, 64);
  build_c2b1<<<cdiv(512L * 256, 256), blk, 0, stream>>>(c2w1, c2b1T);
  build_c2b2<<<cdiv(256L * 128, 256), blk, 0, stream>>>(c2w2, c2b2T);

  // radial MLP (only first 384 outputs are live)
  build_xedge<<<cdiv(E * 192, 256), blk, 0, stream>>>(edist, semb, temb, an, srcI, dstI,
                                                      xeb, E * 192);
  gemmL(xeb, 192, rad1T, rb1, t1, E, 192, 64, 0, 64, 0);
  ln_silu<<<cdiv(E, 8), blk, 0, stream>>>(t1, rg1, rbt1, t1h, (int)E);
  gemmL(t1h, 64, rad2T, rb2, t1, E, 64, 64, 0, 64, 0);
  ln_silu<<<cdiv(E, 8), blk, 0, stream>>>(t1, rg2, rbt2, t1h, (int)E);
  gemmL(t1h, 64, rad3T, rb3, rbuf, E, 64, 384, 1, 384, 0);

  // conv1: only the EXTRA slice survives -> (E,384) @ c1_w0[:, :576]
  build_x0<<<cdiv(E, 8), blk, 0, stream>>>(x, srcI, dstI, wig, rbuf, x0b, (int)E);
  gemmL(x0b, 384, c1w0T, c1b0, x0e, E, 384, 576, 0, 576, 0);

  // gating (-> msg2 in PERM order) + alpha + segment max
  gate_alpha<<<cdiv(E, 8), blk, 0, stream>>>(x0e, x, dstI, rl, alng, alnb, adot,
                                             msg2, alphB, amaxI, (int)E);
  soft_exp<<<cdiv(E * 8, 256), blk, 0, stream>>>(alphB, dstI, amaxI, denB, E * 8);

  // conv2: three WMMA GEMMs writing straight into the final (l,m) row layout
  gemmL(msg2,       576, c2w0T, c2b0,    vmsg, E, 192, 384, 2, 0, 0x620);  // rows 0,2,6
  gemmL(msg2 + 192, 576, c2b1T, nullptr, vmsg, E, 256, 512, 2, 0, 0x5173); // rows 3,7,1,5
  gemmL(msg2 + 448, 576, c2b2T, nullptr, vmsg, E, 128, 256, 2, 0, 0x48);   // rows 8,4

  // attention-weighted Wigner^T rotation + scatter-add
  zero32<<<cdiv(Nn * 1152, 256), blk, 0, stream>>>((unsigned*)node, Nn * 1152);
  attn_scatter<<<cdiv(E, 8), blk, 0, stream>>>(vmsg, alphB, denB, wig, dstI, node, (int)E);

  // final projection: 9 WMMA GEMMs (N,128)@(128,64), weight picked by EXPAND_L
  cvt_f32_f16<<<cdiv(Nn * 1152, 256), blk, 0, stream>>>(node, nodeh, Nn * 1152);
  const int EXP_L[9] = {0, 1, 1, 1, 2, 2, 2, 2, 2};
  for (int i = 0; i < 9; ++i)
    gemmL(nodeh + (long)i * 128, 1152, projT + (long)EXP_L[i] * 8192,
          (i == 0) ? pb : nullptr, outp + (long)i * 64, Nn, 128, 64, 0, 576, 0);
}
